// Model_40630390620627
// MI455X (gfx1250) — compile-verified, matrix operand order checked by reading, not run on hardware
//
#include <hip/hip_runtime.h>
#include <hip/hip_bf16.h>
#include <math.h>

// ---------------------------------------------------------------------------
// Types for CDNA5 WMMA (gfx1250, wave32)
// ---------------------------------------------------------------------------
typedef __attribute__((ext_vector_type(16))) __bf16 v16bf;
typedef __attribute__((ext_vector_type(8)))  float  v8f;

union Frag {
    v16bf v;
    unsigned short u[16];
    uint4 q[2];
};

static __device__ __forceinline__ unsigned short f2bf(float f) {
    unsigned u = __float_as_uint(f);
    u += 0x7FFFu + ((u >> 16) & 1u);          // round-to-nearest-even
    return (unsigned short)(u >> 16);
}

// ---------------------------------------------------------------------------
// Async global->LDS 16B copy (CDNA5 GLOBAL_LOAD_ASYNC_TO_LDS_B128, ASYNCcnt),
// with a synchronous fallback when the builtin isn't available.
// Builtin signature (from hipcc diagnostic): (v4i AS1*, v4i AS3*, imm, imm)
// where v4i = int __attribute__((vector_size(16))).
// ---------------------------------------------------------------------------
#if defined(__HIP_DEVICE_COMPILE__) && defined(__has_builtin)
#  if __has_builtin(__builtin_amdgcn_global_load_async_to_lds_b128)
#    define ASYNC_G2L 1
#  endif
#endif
#ifndef ASYNC_G2L
#  define ASYNC_G2L 0
#endif

typedef int v4i_vs __attribute__((vector_size(16)));

static __device__ __forceinline__ void g2l_16B(const unsigned short* g,
                                               unsigned short* l) {
#if ASYNC_G2L
    typedef __attribute__((address_space(1))) v4i_vs* gvp;
    typedef __attribute__((address_space(3))) v4i_vs* lvp;
    // generic->AS casts via integers: global addresses are numerically equal;
    // LDS generic addresses truncate to the 32-bit LDS offset (ISA 10.2).
    __builtin_amdgcn_global_load_async_to_lds_b128(
        (gvp)(uintptr_t)g, (lvp)(unsigned)(uintptr_t)l, 0, 0);
#else
    *(uint4*)l = *(const uint4*)g;
#endif
}

static __device__ __forceinline__ void g2l_commit() {
#if ASYNC_G2L
    asm volatile("s_wait_asynccnt 0x0" ::: "memory");
#endif
}

// Load a 16x32 bf16 A-style fragment (or B-style fragment of a row-major N x K
// matrix, where "row" is the output column) from LDS.
// Layout (ISA 7.12.2, 16-bit A 16x32): lane half h holds k in {0..7}+8h (regs
// 0..3) and {16..23}+8h (regs 4..7). Caller passes row = <tile row for lane&15>.
static __device__ __forceinline__ void lds_load_frag(const unsigned short* tile,
                                                     int stride, int row, int kofs,
                                                     Frag& f) {
    int half8 = ((threadIdx.x >> 4) & 1) * 8;
    const unsigned short* p = tile + row * stride + kofs + half8;
    f.q[0] = *(const uint4*)(p);
    f.q[1] = *(const uint4*)(p + 16);
}

// B fragment for a row-major K x N matrix in LDS (used for P@V): lane holds
// output column `col`, gathering k = {0..7}+8h and {16..23}+8h down the rows.
static __device__ __forceinline__ void lds_load_frag_col(const unsigned short* tile,
                                                         int stride, int col,
                                                         Frag& f) {
    int half8 = ((threadIdx.x >> 4) & 1) * 8;
#pragma unroll
    for (int i = 0; i < 8; ++i) f.u[i]     = tile[(half8 + i) * stride + col];
#pragma unroll
    for (int i = 0; i < 8; ++i) f.u[8 + i] = tile[(16 + half8 + i) * stride + col];
}

static __device__ __forceinline__ v8f wmma_bf16(const Frag& a, const Frag& b, v8f c) {
    return __builtin_amdgcn_wmma_f32_16x16x32_bf16(false, a.v, false, b.v,
                                                   (short)0, c, false, false);
}

// ---------------------------------------------------------------------------
// Elementwise helpers
// ---------------------------------------------------------------------------
__global__ void convert_f32_bf16(const float* __restrict__ in,
                                 unsigned short* __restrict__ out, int n) {
    int i = blockIdx.x * 256 + threadIdx.x;
    if (i < n) out[i] = f2bf(in[i]);
}

static __device__ __forceinline__ float gelu_tanh(float x) {
    const float c = 0.7978845608028654f;
    return 0.5f * x * (1.0f + tanhf(c * (x + 0.044715f * x * x * x)));
}

// ---------------------------------------------------------------------------
// LayerNorm: one block (256 threads) per row of 768, bf16 output
// ---------------------------------------------------------------------------
__global__ __launch_bounds__(256) void ln_bf16(const float* __restrict__ X,
                                               const float* __restrict__ W,
                                               const float* __restrict__ Bb,
                                               unsigned short* __restrict__ O) {
    const int row = blockIdx.x;
    const int tid = threadIdx.x;
    const float* xr = X + (size_t)row * 768;
    float v0 = xr[tid], v1 = xr[tid + 256], v2 = xr[tid + 512];

    __shared__ float red[8];
    int lane = tid & 31, wv = tid >> 5;

    float s = v0 + v1 + v2;
#pragma unroll
    for (int o = 16; o >= 1; o >>= 1) s += __shfl_xor(s, o, 32);
    if (lane == 0) red[wv] = s;
    __syncthreads();
    float mean = 0.f;
#pragma unroll
    for (int i = 0; i < 8; ++i) mean += red[i];
    mean *= (1.0f / 768.0f);
    __syncthreads();

    float d0 = v0 - mean, d1 = v1 - mean, d2 = v2 - mean;
    s = d0 * d0 + d1 * d1 + d2 * d2;
#pragma unroll
    for (int o = 16; o >= 1; o >>= 1) s += __shfl_xor(s, o, 32);
    if (lane == 0) red[wv] = s;
    __syncthreads();
    float var = 0.f;
#pragma unroll
    for (int i = 0; i < 8; ++i) var += red[i];
    var *= (1.0f / 768.0f);
    float rstd = rsqrtf(var + 1e-5f);

    unsigned short* orow = O + (size_t)row * 768;
    orow[tid]       = f2bf(d0 * rstd * W[tid]       + Bb[tid]);
    orow[tid + 256] = f2bf(d1 * rstd * W[tid + 256] + Bb[tid + 256]);
    orow[tid + 512] = f2bf(d2 * rstd * W[tid + 512] + Bb[tid + 512]);
}

// ---------------------------------------------------------------------------
// Generic WMMA GEMM: out[M,N] = A[M,K](bf16) @ W[N,K]^T(bf16) + bias, epilogues
//   EPI 0: store bf16
//   EPI 1: += residual(fp32), store fp32
//   EPI 2: gelu, store bf16
// Block tile 128x128, 8 waves (4 M x 2 N), per-wave 32x64, K-step 32.
// Double-buffered LDS with async global->LDS fills + L2 prefetch 2 tiles ahead.
// ---------------------------------------------------------------------------
#define LDSTRIDE 40  // 32 + 8 pad, keeps 16B alignment, staggers banks

template <int EPI>
__global__ __launch_bounds__(256) void gemm_bf16(const unsigned short* __restrict__ A,
                                                 const unsigned short* __restrict__ Bw,
                                                 const float* __restrict__ bias,
                                                 const float* __restrict__ residual,
                                                 void* __restrict__ out,
                                                 int M, int N, int K) {
    __shared__ unsigned short sA[2][128 * LDSTRIDE];
    __shared__ unsigned short sB[2][128 * LDSTRIDE];

    const int tid  = threadIdx.x;
    const int wave = tid >> 5;
    const int lane = tid & 31;
    const int l15  = lane & 15;
    const int half = (lane >> 4) & 1;
    const int wm   = wave >> 1;   // 0..3 -> 32-row slab
    const int wn   = wave & 1;    // 0..1 -> 64-col slab
    const int bm   = blockIdx.y * 128;
    const int bn   = blockIdx.x * 128;

    v8f acc[2][4];
#pragma unroll
    for (int i = 0; i < 2; ++i)
#pragma unroll
        for (int j = 0; j < 4; ++j)
#pragma unroll
            for (int e = 0; e < 8; ++e) acc[i][j][e] = 0.0f;

    const int nk = K >> 5;

    auto issue = [&](int kt, int buf) {
        // 128 rows x 32 bf16 = 512 chunks of 16B each, for A and B tiles.
#pragma unroll
        for (int c = tid; c < 512; c += 256) {
            int r = c >> 2, kk = (c & 3) * 8;
            g2l_16B(&A [(size_t)(bm + r) * K + (kt << 5) + kk], &sA[buf][r * LDSTRIDE + kk]);
            g2l_16B(&Bw[(size_t)(bn + r) * K + (kt << 5) + kk], &sB[buf][r * LDSTRIDE + kk]);
        }
    };

    issue(0, 0);
    for (int kt = 0; kt < nk; ++kt) {
        const int buf = kt & 1;
        g2l_commit();          // own wave's async fills for tile kt done
        __syncthreads();       // everyone's fills done + everyone done with buf^1
        if (kt + 1 < nk) issue(kt + 1, buf ^ 1);
        if (kt + 2 < nk) {
            // L2 prefetch two tiles ahead: one 64B row-chunk per thread
            int r = tid & 127;
            const unsigned short* p = (tid < 128)
                ? &A [(size_t)(bm + r) * K + ((kt + 2) << 5)]
                : &Bw[(size_t)(bn + r) * K + ((kt + 2) << 5)];
            __builtin_prefetch(p, 0, 1);
        }

        Frag a[2], b[4];
#pragma unroll
        for (int i = 0; i < 2; ++i)
            lds_load_frag(sA[buf], LDSTRIDE, wm * 32 + i * 16 + l15, 0, a[i]);
#pragma unroll
        for (int j = 0; j < 4; ++j)
            lds_load_frag(sB[buf], LDSTRIDE, wn * 64 + j * 16 + l15, 0, b[j]);
#pragma unroll
        for (int i = 0; i < 2; ++i)
#pragma unroll
            for (int j = 0; j < 4; ++j)
                acc[i][j] = wmma_bf16(a[i], b[j], acc[i][j]);
    }

    // Epilogue. C layout: element e of acc -> row = e + 8*half, col = l15.
#pragma unroll
    for (int i = 0; i < 2; ++i) {
#pragma unroll
        for (int j = 0; j < 4; ++j) {
            int col = bn + wn * 64 + j * 16 + l15;
            float bc = bias[col];
#pragma unroll
            for (int e = 0; e < 8; ++e) {
                int row = bm + wm * 32 + i * 16 + e + 8 * half;
                float val = acc[i][j][e] + bc;
                size_t idx = (size_t)row * N + col;
                if (EPI == 0) {
                    ((unsigned short*)out)[idx] = f2bf(val);
                } else if (EPI == 1) {
                    ((float*)out)[idx] = val + residual[idx];
                } else {
                    ((unsigned short*)out)[idx] = f2bf(gelu_tanh(val));
                }
            }
        }
    }
}

// ---------------------------------------------------------------------------
// Split qkv (M x 2304, bf16) into Q/K/V [B*H, T, 64] bf16
// ---------------------------------------------------------------------------
__global__ void split_qkv(const unsigned short* __restrict__ qkv,
                          unsigned short* __restrict__ Qo,
                          unsigned short* __restrict__ Ko,
                          unsigned short* __restrict__ Vo) {
    size_t i = (size_t)blockIdx.x * 256 + threadIdx.x;  // over 4096*2304
    int m = (int)(i / 2304), n = (int)(i % 2304);
    int which = n / 768, hn = n % 768, hh = hn / 64, d = hn % 64;
    int b = m / 2048, t = m % 2048;
    unsigned short val = qkv[i];
    size_t dst = (((size_t)(b * 12 + hh)) * 2048 + t) * 64 + d;
    unsigned short* o = (which == 0) ? Qo : (which == 1) ? Ko : Vo;
    o[dst] = val;
}

// ---------------------------------------------------------------------------
// Flash attention. grid = (T/64, B*H), 128 threads = 4 waves.
// Each wave owns 16 q rows; K/V streamed in 32-row LDS tiles (async fills);
// causal mask; online softmax with shuffles within each lane half; P@V via WMMA after an
// LDS C-layout -> A-layout transpose.
// ---------------------------------------------------------------------------
__global__ __launch_bounds__(128) void flash_attn(const unsigned short* __restrict__ Q,
                                                  const unsigned short* __restrict__ Kt,
                                                  const unsigned short* __restrict__ Vt,
                                                  unsigned short* __restrict__ Y) {
    __shared__ unsigned short sK[32 * 64];
    __shared__ unsigned short sV[32 * 64];
    __shared__ unsigned short sP[4][16 * 32];

    const int bh   = blockIdx.y;
    const int b    = bh / 12;
    const int h    = bh % 12;
    const int qb   = blockIdx.x * 64;
    const int wave = threadIdx.x >> 5;
    const int lane = threadIdx.x & 31;
    const int l15  = lane & 15;
    const int half = (lane >> 4) & 1;
    const size_t base = (size_t)bh * 2048 * 64;

    // Q fragments for this wave's 16 rows (k-dim 64 -> two 16x32 fragments).
    const int qrow = qb + wave * 16 + l15;
    const unsigned short* qp = Q + base + (size_t)qrow * 64;
    Frag aQ0, aQ1;
    {
        int h8 = half * 8;
        aQ0.q[0] = *(const uint4*)(qp + h8);
        aQ0.q[1] = *(const uint4*)(qp + 16 + h8);
        aQ1.q[0] = *(const uint4*)(qp + 32 + h8);
        aQ1.q[1] = *(const uint4*)(qp + 48 + h8);
    }

    v8f o[4];
#pragma unroll
    for (int j = 0; j < 4; ++j)
#pragma unroll
        for (int e = 0; e < 8; ++e) o[j][e] = 0.0f;
    float m_r[8], l_r[8];
#pragma unroll
    for (int e = 0; e < 8; ++e) { m_r[e] = -INFINITY; l_r[e] = 0.0f; }

    const int row_hi_wave = qb + wave * 16 + 15;
    const int nkt = (qb + 64) / 32;

    for (int kt = 0; kt < nkt; ++kt) {
        __syncthreads();      // previous iteration's reads of sK/sV done
        // Cooperative async fill of 32x64 K and V tiles: 256 16B chunks.
#pragma unroll
        for (int c = threadIdx.x; c < 256; c += 128) {
            int r = c >> 3, d = (c & 7) * 8;
            size_t g = base + (size_t)(kt * 32 + r) * 64 + d;
            g2l_16B(&Kt[g], &sK[r * 64 + d]);
            g2l_16B(&Vt[g], &sV[r * 64 + d]);
        }
        g2l_commit();
        __syncthreads();
        if (kt + 1 < nkt) {
            // L2 prefetch next K/V tile (one 128B row per thread)
            int r = threadIdx.x & 31;
            size_t g = base + (size_t)((kt + 1) * 32 + r) * 64;
            if (threadIdx.x < 32)       __builtin_prefetch(&Kt[g], 0, 1);
            else if (threadIdx.x < 64)  __builtin_prefetch(&Vt[g], 0, 1);
        }

        if (kt * 32 <= row_hi_wave) {   // wave-uniform: EXEC stays all-ones
            // ---- S = (Q @ K^T) * 1/sqrt(64), two 16x16 column chunks ----
            v8f s[2];
#pragma unroll
            for (int c = 0; c < 2; ++c) {
                Frag bK0, bK1;
                lds_load_frag(sK, 64, c * 16 + l15, 0,  bK0);
                lds_load_frag(sK, 64, c * 16 + l15, 32, bK1);
                v8f z;
#pragma unroll
                for (int e = 0; e < 8; ++e) z[e] = 0.0f;
                s[c] = wmma_bf16(aQ0, bK0, z);
                s[c] = wmma_bf16(aQ1, bK1, s[c]);
            }
            const int row_lane_base = qb + wave * 16 + 8 * half;
#pragma unroll
            for (int c = 0; c < 2; ++c) {
                int kcol = kt * 32 + c * 16 + l15;
#pragma unroll
                for (int e = 0; e < 8; ++e) {
                    float sv = s[c][e] * 0.125f;
                    if (kcol > row_lane_base + e) sv = -INFINITY;
                    s[c][e] = sv;
                }
            }
            // ---- online softmax (per row e, reduce across 16 lanes of half) ----
            float alpha[8];
#pragma unroll
            for (int e = 0; e < 8; ++e) {
                float rm = fmaxf(s[0][e], s[1][e]);
#pragma unroll
                for (int off = 8; off >= 1; off >>= 1)
                    rm = fmaxf(rm, __shfl_xor(rm, off, 32));
                float nm = fmaxf(m_r[e], rm);
                alpha[e] = __expf(m_r[e] - nm);   // first iter: exp(-inf)=0
                m_r[e] = nm;
                float p0 = __expf(s[0][e] - nm);
                float p1 = __expf(s[1][e] - nm);
                s[0][e] = p0; s[1][e] = p1;
                float rs = p0 + p1;
#pragma unroll
                for (int off = 8; off >= 1; off >>= 1)
                    rs += __shfl_xor(rs, off, 32);
                l_r[e] = l_r[e] * alpha[e] + rs;
            }
            // ---- transpose P (C-layout) -> A-layout via per-wave LDS ----
            unsigned short* pw = &sP[wave][0];
#pragma unroll
            for (int c = 0; c < 2; ++c)
#pragma unroll
                for (int e = 0; e < 8; ++e)
                    pw[(e + 8 * half) * 32 + c * 16 + l15] = f2bf(s[c][e]);
            asm volatile("s_wait_dscnt 0x0" ::: "memory");
            Frag aP;
            lds_load_frag(pw, 32, l15, 0, aP);
            // ---- rescale O and accumulate P @ V ----
#pragma unroll
            for (int j = 0; j < 4; ++j)
#pragma unroll
                for (int e = 0; e < 8; ++e) o[j][e] *= alpha[e];
#pragma unroll
            for (int j = 0; j < 4; ++j) {
                Frag bV;
                lds_load_frag_col(sV, 64, j * 16 + l15, bV);
                o[j] = wmma_bf16(aP, bV, o[j]);
            }
        }
    }

    // ---- normalize and write y[b, t, h*64 + d] as bf16 ----
#pragma unroll
    for (int j = 0; j < 4; ++j) {
#pragma unroll
        for (int e = 0; e < 8; ++e) {
            int t = qb + wave * 16 + e + 8 * half;
            int d = j * 16 + l15;
            float val = o[j][e] / l_r[e];
            Y[((size_t)(b * 2048 + t)) * 768 + h * 64 + d] = f2bf(val);
        }
    }
}

// ---------------------------------------------------------------------------
// Host orchestration
// ---------------------------------------------------------------------------
extern "C" void kernel_launch(void* const* d_in, const int* in_sizes, int n_in,
                              void* d_out, int out_size, void* d_ws, size_t ws_size,
                              hipStream_t stream) {
    (void)in_sizes; (void)n_in; (void)out_size; (void)ws_size;

    const int B = 2, T = 2048, C = 768, H = 12;
    const int M = B * T;                 // 4096
    const size_t C3 = 3 * C, C4 = 4 * C;

    const float* x        = (const float*)d_in[0];
    const float* ln1_w    = (const float*)d_in[1];
    const float* ln1_b    = (const float*)d_in[2];
    const float* wqkv     = (const float*)d_in[3];
    const float* bqkv     = (const float*)d_in[4];
    const float* wproj    = (const float*)d_in[5];
    const float* bproj    = (const float*)d_in[6];
    const float* ln2_w    = (const float*)d_in[7];
    const float* ln2_b    = (const float*)d_in[8];
    const float* wfc      = (const float*)d_in[9];
    const float* bfc      = (const float*)d_in[10];
    const float* wproj2   = (const float*)d_in[11];
    const float* bproj2   = (const float*)d_in[12];

    // workspace carve-up (256B aligned)
    char* ws = (char*)d_ws;
    size_t off = 0;
    auto take = [&](size_t bytes) {
        char* p = ws + off;
        off += (bytes + 255) & ~(size_t)255;
        return p;
    };
    unsigned short* WQKVb  = (unsigned short*)take(C3 * C * 2);
    unsigned short* WPROJb = (unsigned short*)take((size_t)C * C * 2);
    unsigned short* WFCb   = (unsigned short*)take(C4 * C * 2);
    unsigned short* WPR2b  = (unsigned short*)take((size_t)C * C4 * 2);
    unsigned short* H1     = (unsigned short*)take((size_t)M * C * 2);
    unsigned short* QKV    = (unsigned short*)take((size_t)M * C3 * 2);
    unsigned short* Qb     = (unsigned short*)take((size_t)B * H * T * 64 * 2);
    unsigned short* Kb     = (unsigned short*)take((size_t)B * H * T * 64 * 2);
    unsigned short* Vb     = (unsigned short*)take((size_t)B * H * T * 64 * 2);
    unsigned short* Yb     = (unsigned short*)take((size_t)M * C * 2);
    float*          X2     = (float*)take((size_t)M * C * 4);
    unsigned short* H2     = (unsigned short*)take((size_t)M * C * 2);
    unsigned short* ACT    = (unsigned short*)take((size_t)M * C4 * 2);

    // 1) weights -> bf16
    {
        int n1 = (int)(C3 * C), n2 = C * C, n3 = (int)(C4 * C), n4 = (int)(C * C4);
        convert_f32_bf16<<<(n1 + 255) / 256, 256, 0, stream>>>(wqkv,   WQKVb,  n1);
        convert_f32_bf16<<<(n2 + 255) / 256, 256, 0, stream>>>(wproj,  WPROJb, n2);
        convert_f32_bf16<<<(n3 + 255) / 256, 256, 0, stream>>>(wfc,    WFCb,   n3);
        convert_f32_bf16<<<(n4 + 255) / 256, 256, 0, stream>>>(wproj2, WPR2b,  n4);
    }
    // 2) ln1
    ln_bf16<<<M, 256, 0, stream>>>(x, ln1_w, ln1_b, H1);
    // 3) qkv = h1 @ Wqkv^T + b
    gemm_bf16<0><<<dim3((int)C3 / 128, M / 128), 256, 0, stream>>>(
        H1, WQKVb, bqkv, nullptr, QKV, M, (int)C3, C);
    // 4) split into per-head Q/K/V
    {
        size_t n = (size_t)M * C3;
        split_qkv<<<(unsigned)(n / 256), 256, 0, stream>>>(QKV, Qb, Kb, Vb);
    }
    // 5) causal flash attention -> Yb [B,T,C] bf16
    flash_attn<<<dim3(T / 64, B * H), 128, 0, stream>>>(Qb, Kb, Vb, Yb);
    // 6) x2 = x + y @ Wproj^T + b   (fp32)
    gemm_bf16<1><<<dim3(C / 128, M / 128), 256, 0, stream>>>(
        Yb, WPROJb, bproj, x, X2, M, C, C);
    // 7) ln2
    ln_bf16<<<M, 256, 0, stream>>>(X2, ln2_w, ln2_b, H2);
    // 8) act = gelu(h2 @ Wfc^T + b)  (bf16)
    gemm_bf16<2><<<dim3((int)C4 / 128, M / 128), 256, 0, stream>>>(
        H2, WFCb, bfc, nullptr, ACT, M, (int)C4, C);
    // 9) out = x2 + act @ Wproj2^T + b  (fp32)
    gemm_bf16<1><<<dim3(C / 128, M / 128), 256, 0, stream>>>(
        ACT, WPR2b, bproj2, X2, d_out, M, C, (int)C4);
}
